// EmbedTargetLocMulti_44667659879067
// MI455X (gfx1250) — compile-verified, matrix operand order checked by reading, not run on hardware
//
#include <hip/hip_runtime.h>
#include <hip/hip_bf16.h>

typedef __attribute__((ext_vector_type(16))) __bf16 v16bf;
typedef __attribute__((ext_vector_type(8)))  __bf16 v8bf;
typedef __attribute__((ext_vector_type(8)))  float  v8f;

#define B_TOTAL 32768
#define NJ      24
#define DD      128
#define TILE_M  128
#define LDA     136   // bf16 elems/row for 128-wide tiles: 272B stride (16B pad -> bank rotate)
#define LDB2    40    // bf16 elems/row for 32-wide tiles: 80B stride

__device__ __forceinline__ v16bf load_frag(const __bf16* p) {
    const v8bf lo = *(const v8bf*)p;
    const v8bf hi = *(const v8bf*)(p + 16);
    return __builtin_shufflevector(lo, hi, 0,1,2,3,4,5,6,7,8,9,10,11,12,13,14,15);
}

__global__ __launch_bounds__(256)
void embed_target_loc_wmma(const float* __restrict__ X,      // [B, J, 3]
                           const int*   __restrict__ jmask,  // [B, J]
                           const int*   __restrict__ thead,  // [B]
                           const float* __restrict__ W1,     // [J, 3, D]
                           const float* __restrict__ b1,     // [J, D]
                           const float* __restrict__ W2,     // [J, D, D]
                           const float* __restrict__ b2,     // [J, D]
                           const float* __restrict__ sum_w,  // [J]
                           float*       __restrict__ out)    // [B, D]
{
    __shared__ __attribute__((aligned(16))) __bf16 As   [TILE_M * LDA];  // h' tile (A)
    __shared__ __attribute__((aligned(16))) __bf16 Bs   [DD * LDA];      // W2[j]^T (B, [n][k])
    __shared__ __attribute__((aligned(16))) __bf16 Ss   [TILE_M * 32];   // scales, J padded to 32
    __shared__ __attribute__((aligned(16))) __bf16 B2T  [DD * LDB2];     // b2^T, [n][j]
    __shared__ __attribute__((aligned(16))) float  W1all[NJ * 3 * DD];   // all joint W1
    __shared__ __attribute__((aligned(16))) float  b1all[NJ * DD];       // all joint b1
    __shared__ __attribute__((aligned(16))) float  Xall [TILE_M * NJ * 3]; // x tile, 72 f32/row

    const int tid  = threadIdx.x;
    const int row0 = blockIdx.x * TILE_M;

    // ---------- prologue (all staged once) ----------
    if (tid < TILE_M) {
        const int b  = row0 + tid;
        const int th = thead[b];
        #pragma unroll
        for (int j = 0; j < 32; ++j) {
            float v = 0.0f;
            if (j < NJ) {
                const int m = (j == NJ - 1) ? th : jmask[(size_t)b * NJ + j];
                if (m != 0) v = sum_w[j];
            }
            Ss[tid * 32 + j] = (__bf16)v;
        }
    }
    for (int i = tid; i < DD * 32; i += 256) {          // b2^T, joints padded to 32
        const int n = i >> 5, k = i & 31;
        B2T[n * LDB2 + k] = (__bf16)((k < NJ) ? b2[(size_t)k * DD + n] : 0.0f);
    }
    for (int i = tid; i < NJ * 3 * DD; i += 256) W1all[i] = W1[i];
    for (int i = tid; i < NJ * DD;     i += 256) b1all[i] = b1[i];
    {   // x tile: contiguous [128 rows x 72 f32] -> float4 loads
        const float4* Xg = (const float4*)(X + (size_t)row0 * NJ * 3);
        float4*       Xl = (float4*)Xall;
        for (int i = tid; i < TILE_M * NJ * 3 / 4; i += 256) Xl[i] = Xg[i];
    }
    __syncthreads();

    // wave -> 32-row x 64-col output slice: each B fragment feeds TWO WMMAs
    const int wave  = tid >> 5;          // 8 wave32s
    const int lane  = tid & 31;
    const int half  = lane >> 4;
    const int l16   = lane & 15;
    const int mrow  = (wave & 3) * 32;   // M group (32 rows)
    const int ncol0 = (wave >> 2) * 64;  // N group (4 tiles of 16)
    const int rowLo = mrow + l16;        // A-frag row, low 16
    const int rowHi = mrow + 16 + l16;   // A-frag row, high 16
    const int off1  = half * 8;          // K sub-offset per documented 16-bit A layout

    v8f accL[4] = {};                    // rows [mrow, mrow+16)
    v8f accH[4] = {};                    // rows [mrow+16, mrow+32)

    for (int j = 0; j < NJ; ++j) {
        // ---- phase 1: stage W2[j]^T -> Bs (f32 b128 loads), h' -> As; overlapped
        const float* W2j = W2 + (size_t)j * DD * DD;
        #pragma unroll 4
        for (int i = 0; i < 16; ++i) {
            const int f4 = tid + 256 * i;            // float4 index: 0..4095
            const float4 v = ((const float4*)W2j)[f4];
            const int k = f4 >> 5;                   // row (K) of W2
            const int n = (f4 & 31) * 4;             // col (N) of W2
            Bs[(n + 0) * LDA + k] = (__bf16)v.x;
            Bs[(n + 1) * LDA + k] = (__bf16)v.y;
            Bs[(n + 2) * LDA + k] = (__bf16)v.z;
            Bs[(n + 3) * LDA + k] = (__bf16)v.w;
        }
        {   // h'[r][d] = s[r,j] * silu(x . W1[j,:,d] + b1[j,d]) -> As (bf16)
            const int r  = tid >> 1;
            const int c0 = (tid & 1) * 64;
            const float x0 = Xall[r * 72 + j * 3 + 0];
            const float x1 = Xall[r * 72 + j * 3 + 1];
            const float x2 = Xall[r * 72 + j * 3 + 2];
            const float s  = (float)Ss[r * 32 + j];
            const float* w1j = &W1all[j * 3 * DD];
            const float* b1j = &b1all[j * DD];
            #pragma unroll 8
            for (int c = 0; c < 64; ++c) {
                const int dd = c0 + c;
                const float z  = fmaf(x0, w1j[dd],
                                 fmaf(x1, w1j[DD + dd],
                                 fmaf(x2, w1j[2 * DD + dd], b1j[dd])));
                const float hv = s * (z / (1.0f + __expf(-z)));   // s * silu(z)
                As[r * LDA + dd] = (__bf16)hv;
            }
        }
        if (j + 1 < NJ)
            __builtin_prefetch(W2 + (size_t)(j + 1) * DD * DD + tid * 64, 0, 1);
        __syncthreads();

        // ---- phase 2: 32x64 slice of h' @ W2[j]; 2 WMMAs per B fragment by reuse
        #pragma unroll
        for (int kk = 0; kk < DD; kk += 32) {
            const v16bf aL = load_frag(&As[rowLo * LDA + kk + off1]);
            const v16bf aH = load_frag(&As[rowHi * LDA + kk + off1]);
            #pragma unroll
            for (int np = 0; np < 2; ++np) {         // pairs of N-tiles
                const int nt0 = np * 2, nt1 = np * 2 + 1;
                const v16bf b0 = load_frag(&Bs[(ncol0 + nt0 * 16 + l16) * LDA + kk + off1]);
                const v16bf b1f = load_frag(&Bs[(ncol0 + nt1 * 16 + l16) * LDA + kk + off1]);
                accL[nt0] = __builtin_amdgcn_wmma_f32_16x16x32_bf16(
                    false, aL, false, b0, (short)0, accL[nt0], false, false);
                accH[nt0] = __builtin_amdgcn_wmma_f32_16x16x32_bf16(
                    false, aH, false, b0, (short)0, accH[nt0], false, false);
                accL[nt1] = __builtin_amdgcn_wmma_f32_16x16x32_bf16(
                    false, aL, false, b1f, (short)0, accL[nt1], false, false);
                accH[nt1] = __builtin_amdgcn_wmma_f32_16x16x32_bf16(
                    false, aH, false, b1f, (short)0, accH[nt1], false, false);
            }
        }
        __syncthreads();   // before next j reuses As/Bs
    }

    // ---- bias epilogue: acc += S(128x32) @ b2^T(32x128); K=32, B frags reused 2x
    {
        const v16bf aL = load_frag(&Ss[rowLo * 32 + off1]);
        const v16bf aH = load_frag(&Ss[rowHi * 32 + off1]);
        #pragma unroll
        for (int nt = 0; nt < 4; ++nt) {
            const v16bf bb = load_frag(&B2T[(ncol0 + nt * 16 + l16) * LDB2 + off1]);
            accL[nt] = __builtin_amdgcn_wmma_f32_16x16x32_bf16(
                false, aL, false, bb, (short)0, accL[nt], false, false);
            accH[nt] = __builtin_amdgcn_wmma_f32_16x16x32_bf16(
                false, aH, false, bb, (short)0, accH[nt], false, false);
        }
    }

    // ---- store C fragments: VGPR v holds M = v + 8*half, N = l16 within each tile
    #pragma unroll
    for (int nt = 0; nt < 4; ++nt) {
        const int colg = ncol0 + nt * 16 + l16;
        #pragma unroll
        for (int v = 0; v < 8; ++v) {
            const int rrL = mrow + half * 8 + v;
            out[(size_t)(row0 + rrL) * DD + colg] = accL[nt][v];
        }
        #pragma unroll
        for (int v = 0; v < 8; ++v) {
            const int rrH = mrow + 16 + half * 8 + v;
            out[(size_t)(row0 + rrH) * DD + colg] = accH[nt][v];
        }
    }
}

extern "C" void kernel_launch(void* const* d_in, const int* in_sizes, int n_in,
                              void* d_out, int out_size, void* d_ws, size_t ws_size,
                              hipStream_t stream) {
    const float* X  = (const float*)d_in[0];
    const int*   jm = (const int*)  d_in[1];
    const int*   th = (const int*)  d_in[2];
    const float* W1 = (const float*)d_in[3];
    const float* b1 = (const float*)d_in[4];
    const float* W2 = (const float*)d_in[5];
    const float* b2 = (const float*)d_in[6];
    const float* sw = (const float*)d_in[7];
    float*       out = (float*)d_out;

    dim3 grid(B_TOTAL / TILE_M);   // 256 blocks
    dim3 block(256);               // 8 wave32s
    hipLaunchKernelGGL(embed_target_loc_wmma, grid, block, 0, stream,
                       X, jm, th, W1, b1, W2, b2, sw, out);
}
